// CategorySpecificLinear_82471962017991
// MI455X (gfx1250) — matrix operand order repaired and checked by reading, not compiled, not run on hardware
//
#include <hip/hip_runtime.h>

// ---------------------------------------------------------------------------
// CategorySpecificLinear for MI455X (gfx1250, wave32, WMMA).
// out[b,s,h] = sum_d x[b,s,d] * W[cat[b],d,h] + bias[cat[b],h]
// B=256, S=64, D=1024, H=1024, fp32 in/out, bf16 WMMA with f32 accumulate.
// Memory-bound (~384MB min traffic @ 23.3TB/s ~ 16.5us); bf16 WMMA keeps
// compute (~7us at dense-bf16 matrix rates) below the memory floor.
// ---------------------------------------------------------------------------

typedef __attribute__((ext_vector_type(16))) __bf16        v16bf;
typedef __attribute__((ext_vector_type(8)))  float         v8f;
typedef __attribute__((ext_vector_type(4)))  unsigned int  v4u;
typedef __attribute__((ext_vector_type(8)))  unsigned int  v8u;

#define BATCH       256
#define SEQ         64
#define IN_DIM      1024
#define HID_DIM     1024

#define KT          32                 // K per WMMA step
#define NT          256                // N tile per block
#define KSTEPS      (IN_DIM / KT)      // 32
#define A_STRIDE    40                 // ushorts/row: 80B (16B aligned, 20-bank skew)
#define B_STRIDE    40

// fp32x2 -> packed bf16x2 in 3 VALU: round-half-up then v_perm_b32 byte-select
// of the two high halves. (RNE would cost ~10 VALU/pair; tie-rounding delta is
// irrelevant here and v_perm is target-generic, unlike cvt_pk_bf16 builtins.)
__device__ __forceinline__ unsigned int pack2(float lo, float hi) {
  const unsigned int lr = __builtin_bit_cast(unsigned int, lo) + 0x8000u;
  const unsigned int hr = __builtin_bit_cast(unsigned int, hi) + 0x8000u;
  // result bytes: {hr[3], hr[2], lr[3], lr[2]}  (src0=hr -> sel 7..4, src1=lr -> sel 3..0)
  return __builtin_amdgcn_perm(hr, lr, 0x07060302u);
}

__global__ __launch_bounds__(256)
void cat_linear_wmma(const float* __restrict__ x,
                     const int*   __restrict__ cat_ids,
                     const float* __restrict__ W,
                     const float* __restrict__ bias,
                     float*       __restrict__ out) {
  // Double-buffered LDS staging (bf16): x tile 64x32, W tile transposed 256x32
  __shared__ unsigned short Abuf[2][SEQ * A_STRIDE];   // [s][k]
  __shared__ unsigned short Bbuf[2][NT  * B_STRIDE];   // [h][k]  (transposed)

  const int tid  = threadIdx.x;
  const int bidx = blockIdx.x;          // sample
  const int h0   = blockIdx.y * NT;     // H tile base
  const int e    = cat_ids[bidx];       // expert id

  const float* xb = x + (size_t)bidx * SEQ * IN_DIM;
  const float* We = W + (size_t)e * IN_DIM * HID_DIM;

  const int wave = tid >> 5;
  const int lane = tid & 31;
  const int half = lane >> 4;           // 0: lanes 0-15, 1: lanes 16-31
  const int lcol = lane & 15;
  const int m0   = (wave & 3) * 16;     // wave's S-row tile (4 waves cover 64)
  const int wn0  = (wave >> 2) * 128;   // wave's N tile within NT (2 waves)

  auto stage = [&](int kb, int buf) {
    const int k0 = kb * KT;
    // ---- x tile: SEQ x KT fp32 -> bf16 row-major. 512 float4 / 256 thr.
    {
      int f = tid;
#pragma unroll
      for (int it = 0; it < 2; ++it, f += 256) {
        const int s = f >> 3;           // 0..63
        const int q = f & 7;            // float4 index within row
        const float4 v = *(const float4*)(xb + (size_t)s * IN_DIM + k0 + q * 4);
        unsigned int* dst = (unsigned int*)&Abuf[buf][s * A_STRIDE + q * 4];
        dst[0] = pack2(v.x, v.y);
        dst[1] = pack2(v.z, v.w);
      }
    }
    // ---- W tile: KT x NT fp32, stored transposed [h][k]. Each thread owns
    // one h-column (coalesced b32 loads across lanes), packs K-pairs.
    {
      const float*     wcol = We + (size_t)k0 * HID_DIM + h0 + tid;
      unsigned short*  dcol = &Bbuf[buf][tid * B_STRIDE];
#pragma unroll
      for (int kp = 0; kp < KT / 2; ++kp) {
        const float w0 = wcol[(size_t)(2 * kp)     * HID_DIM];
        const float w1 = wcol[(size_t)(2 * kp + 1) * HID_DIM];
        *(unsigned int*)&dcol[2 * kp] = pack2(w0, w1);
      }
    }
  };

  v8f acc[8] = {};

  stage(0, 0);
  __syncthreads();

  for (int kb = 0; kb < KSTEPS; ++kb) {
    const int buf = kb & 1;
    if (kb + 1 < KSTEPS) stage(kb + 1, buf ^ 1);   // prefetch next K slice

    // A fragment (16x32 bf16): lane m holds K = {half*8..+7, 16+half*8..+7}
    const unsigned short* arow = &Abuf[buf][(m0 + lcol) * A_STRIDE + half * 8];
    v8u au;
    au.lo = *(const v4u*)(arow);        // ds_load_b128: K = half*8 .. +7
    au.hi = *(const v4u*)(arow + 16);   // ds_load_b128: K = 16+half*8 .. +7
    const v16bf afrag = __builtin_bit_cast(v16bf, au);

#pragma unroll
    for (int j = 0; j < 8; ++j) {
      // B fragment (32x16 bf16): lane n holds K = {half*8..+7, 16+half*8..+7}
      const unsigned short* brow =
          &Bbuf[buf][(wn0 + j * 16 + lcol) * B_STRIDE + half * 8];
      v8u bu;
      bu.lo = *(const v4u*)(brow);      // e = 0..7
      bu.hi = *(const v4u*)(brow + 16); // e = 8..15
      const v16bf bfrag = __builtin_bit_cast(v16bf, bu);

      acc[j] = __builtin_amdgcn_wmma_f32_16x16x32_bf16(
          /*neg_a=*/false, afrag, /*neg_b=*/false, bfrag,
          /*c_mod=*/(short)0, acc[j], /*reuse_a=*/false, /*reuse_b=*/false);
    }
    __syncthreads();
  }

  // ---- epilogue: D VGPR r -> row half*8 + r, col = lane&15 within subtile
  const float* bp   = bias + (size_t)e * HID_DIM;
  const int    row0 = m0 + half * 8;
#pragma unroll
  for (int j = 0; j < 8; ++j) {
    const int   col = h0 + wn0 + j * 16 + lcol;
    const float bj  = bp[col];
    float* o = out + ((size_t)bidx * SEQ + row0) * HID_DIM + col;
#pragma unroll
    for (int r = 0; r < 8; ++r) {
      o[(size_t)r * HID_DIM] = acc[j][r] + bj;
    }
  }
}

extern "C" void kernel_launch(void* const* d_in, const int* in_sizes, int n_in,
                              void* d_out, int out_size, void* d_ws, size_t ws_size,
                              hipStream_t stream) {
  (void)in_sizes; (void)n_in; (void)out_size; (void)d_ws; (void)ws_size;
  const float* x       = (const float*)d_in[0];
  const int*   cat_ids = (const int*)  d_in[1];
  const float* W       = (const float*)d_in[2];
  const float* bias    = (const float*)d_in[3];
  float*       out     = (float*)d_out;

  dim3 grid(BATCH, HID_DIM / NT, 1);   // 256 x 4 blocks
  dim3 block(256, 1, 1);               // 8 wave32 per block
  cat_linear_wmma<<<grid, block, 0, stream>>>(x, cat_ids, W, bias, out);
}
